// SDT_85813446574128
// MI455X (gfx1250) — matrix operand order repaired and checked by reading, not compiled, not run on hardware
//
#include <hip/hip_runtime.h>
#include <cmath>

// ---------------------------------------------------------------------------
// Soft Decision Tree forward for MI455X (gfx1250, wave32, WMMA bf16).
//
// Heavy ops = two GEMMs (8192x2049x2047 and 8192x1024x1000) -> v_wmma_f32_16x16x32_bf16.
// f32 inputs are converted once to bf16; accumulation stays f32; epilogues
// (bias+sigmoid / store guard) run in f32.  Everything else is bandwidth-bound
// elementwise / reduction work sized for 23.3 TB/s HBM.
// ---------------------------------------------------------------------------

typedef __bf16 bf16;
typedef __attribute__((ext_vector_type(16))) __bf16 v16bf;
typedef __attribute__((ext_vector_type(8)))  __bf16 v8bf;
typedef __attribute__((ext_vector_type(8)))  float  v8f;

#define SDT_BATCH     8192
#define SDT_INDIM     2048
#define SDT_KPAD1     2048   // GEMM1 K (X columns; bias handled separately)
#define SDT_NNODES    2047   // inner nodes
#define SDT_NPAD1     2048   // padded N for GEMM1
#define SDT_NLEAF     1024
#define SDT_OUTDIM    1000
#define SDT_NPAD2     1024   // padded N for GEMM2

// ---------------------- conversion kernels ---------------------------------

__global__ void sdt_conv_x(const float* __restrict__ x, bf16* __restrict__ o, int n) {
    int i = blockIdx.x * blockDim.x + threadIdx.x;
    if (i < n) o[i] = (bf16)x[i];
}

// W_inner (2047 x 2049, col0 = bias) -> Wt bf16 (2048 x 2048, zero padded) + bias f32 (2048)
__global__ void sdt_conv_winner(const float* __restrict__ w,
                                bf16* __restrict__ wt, float* __restrict__ bias) {
    int i = blockIdx.x * blockDim.x + threadIdx.x;       // 2048*2048 threads
    int nrow = i >> 11;
    int k    = i & 2047;
    float v = (nrow < SDT_NNODES) ? w[(size_t)nrow * (SDT_INDIM + 1) + 1 + k] : 0.0f;
    wt[i] = (bf16)v;
    if (k == 0) bias[nrow] = (nrow < SDT_NNODES) ? w[(size_t)nrow * (SDT_INDIM + 1)] : 0.0f;
}

// W_leaf (1000 x 1024) -> bf16 (1024 x 1024, zero padded rows)
__global__ void sdt_conv_wleaf(const float* __restrict__ w, bf16* __restrict__ o) {
    int i = blockIdx.x * blockDim.x + threadIdx.x;       // 1024*1024 threads
    int nrow = i >> 10;
    int k    = i & 1023;
    o[i] = (bf16)((nrow < SDT_OUTDIM) ? w[(size_t)nrow * SDT_NLEAF + k] : 0.0f);
}

// ---------------------- WMMA bf16 GEMM (C = A * Bw^T [+bias][sigmoid]) -----
// A: (M x K) bf16 row-major, Bw: (Npad x K) bf16 row-major (zero padded).
// Block tile: 64(M) x 128(N), 8 wave32 waves in 2(M) x 4(N), wave tile 32x32
// = 2x2 fragments of v_wmma_f32_16x16x32_bf16.  K staged through LDS
// (stride padded to 40 bf16 = 80B to spread bank accesses).

template <bool SIGMOID>
__global__ __launch_bounds__(256)
void sdt_gemm_bf16(const bf16* __restrict__ A, const bf16* __restrict__ Bw,
                   const float* __restrict__ bias, float* __restrict__ C,
                   int K, int ldc, int nvalid) {
    constexpr int LDS_STRIDE = 40;                       // bf16 elems per staged row
    __shared__ bf16 Ash[64  * LDS_STRIDE];               // 5.0 KB
    __shared__ bf16 Bsh[128 * LDS_STRIDE];               // 10.0 KB

    const int tid  = threadIdx.x;
    const int lane = tid & 31;
    const int wid  = tid >> 5;
    const int wn   = wid & 3;        // wave N position (0..3)
    const int wm   = wid >> 2;       // wave M position (0..1)
    const int lrow = lane & 15;      // fragment row/col within 16
    const int hv   = lane >> 4;      // lane half (0/1)

    const int mblock = blockIdx.y * 64;
    const int nblock = blockIdx.x * 128;

    v8f acc[2][2] = {};

    // staging assignments: A tile = 64 rows x 32 bf16 (256 x 16B chunks),
    // B tile = 128 rows x 32 bf16 (512 chunks -> 2 per thread).
    const int arow = tid >> 2, ac = tid & 3;
    const bf16* aG = A + (size_t)(mblock + arow) * K + ac * 8;
    bf16* aS = &Ash[arow * LDS_STRIDE + ac * 8];

    const int br0 = tid >> 2,          bc0 = tid & 3;
    const int br1 = (tid + 256) >> 2,  bc1 = tid & 3;
    const bf16* bG0 = Bw + (size_t)(nblock + br0) * K + bc0 * 8;
    const bf16* bG1 = Bw + (size_t)(nblock + br1) * K + bc1 * 8;
    bf16* bS0 = &Bsh[br0 * LDS_STRIDE + bc0 * 8];
    bf16* bS1 = &Bsh[br1 * LDS_STRIDE + bc1 * 8];

    union Frag { v16bf v; v8bf h[2]; };

    for (int k0 = 0; k0 < K; k0 += 32) {
        // stage tiles to LDS (b128 moves)
        *(v8bf*)aS  = *(const v8bf*)(aG  + k0);
        *(v8bf*)bS0 = *(const v8bf*)(bG0 + k0);
        *(v8bf*)bS1 = *(const v8bf*)(bG1 + k0);
        if (k0 + 64 < K) {                               // hint next-next K tile
            __builtin_prefetch(aG  + k0 + 64, 0, 1);
            __builtin_prefetch(bG0 + k0 + 64, 0, 1);
            __builtin_prefetch(bG1 + k0 + 64, 0, 1);
        }
        __syncthreads();

        Frag af[2], bf[2];
#pragma unroll
        for (int f = 0; f < 2; ++f) {
            // A lane layout (16-bit A 16x32): lanes 0-15 hold K {0..7,16..23},
            // lanes 16-31 hold K {8..15,24..31}  -> chunks at 8*hv and 16+8*hv
            int ra = (wm * 32 + f * 16 + lrow) * LDS_STRIDE;
            af[f].h[0] = *(const v8bf*)&Ash[ra + 8 * hv];
            af[f].h[1] = *(const v8bf*)&Ash[ra + 16 + 8 * hv];
            // B lane layout (32x16): same K order as A -> chunks at 8*hv and 16+8*hv
            int rb = (wn * 32 + f * 16 + lrow) * LDS_STRIDE;
            bf[f].h[0] = *(const v8bf*)&Bsh[rb + 8 * hv];
            bf[f].h[1] = *(const v8bf*)&Bsh[rb + 16 + 8 * hv];
        }
#pragma unroll
        for (int i = 0; i < 2; ++i)
#pragma unroll
            for (int j = 0; j < 2; ++j)
                acc[i][j] = __builtin_amdgcn_wmma_f32_16x16x32_bf16(
                    false, af[i].v, false, bf[j].v,
                    (short)0, acc[i][j], false, false);
        __syncthreads();
    }

    // epilogue: C/D layout -> lane half selects M block of 8, VGPR r = row.
    float bv[2] = {0.0f, 0.0f};
    if (bias) {
#pragma unroll
        for (int j = 0; j < 2; ++j)
            bv[j] = bias[nblock + wn * 32 + j * 16 + lrow];
    }
#pragma unroll
    for (int i = 0; i < 2; ++i) {
#pragma unroll
        for (int j = 0; j < 2; ++j) {
            int n = nblock + wn * 32 + j * 16 + lrow;
            if (n >= nvalid) continue;
#pragma unroll
            for (int r = 0; r < 8; ++r) {
                int m = mblock + wm * 32 + i * 16 + hv * 8 + r;
                float v = acc[i][j][r] + bv[j];
                if (SIGMOID) v = 1.0f / (1.0f + __expf(-v));
                C[(size_t)m * ldc + n] = v;
            }
        }
    }
}

// ---------------------- mu cascade (one block per batch row) ---------------

__global__ __launch_bounds__(128)
void sdt_mu(const float* __restrict__ P, int ldp,
            float* __restrict__ allpp, float* __restrict__ muOut,
            bf16* __restrict__ muBf) {
    __shared__ float pr[SDT_NNODES];
    __shared__ float mu[SDT_NNODES];
    const int b = blockIdx.x;
    const int t = threadIdx.x;
    for (int i = t; i < SDT_NNODES; i += 128) pr[i] = P[(size_t)b * ldp + i];
    if (t == 0) mu[0] = 1.0f;
    __syncthreads();
#pragma unroll
    for (int d = 0; d < 10; ++d) {
        int n = 1 << d, off = n - 1;
        for (int j = t; j < n; j += 128) {
            float m  = mu[off + j];
            float pv = pr[off + j];
            mu[2 * off + 1 + 2 * j] = m * (1.0f - pv);   // left child
            mu[2 * off + 2 + 2 * j] = m * pv;            // right child
        }
        __syncthreads();
    }
    for (int i = t; i < SDT_NNODES; i += 128)
        allpp[(size_t)b * SDT_NNODES + i] = mu[i];
    for (int i = t; i < SDT_NLEAF; i += 128) {
        float v = mu[SDT_NLEAF - 1 + i];
        muOut[(size_t)b * SDT_NLEAF + i] = v;
        muBf[(size_t)b * SDT_NLEAF + i] = (bf16)v;
    }
}

// ---------------------- penalty (deterministic two-stage) ------------------

__global__ void sdt_pen_cols(const float* __restrict__ P, int ldp,
                             const float* __restrict__ allpp,
                             float* __restrict__ pencol) {
    int k = blockIdx.x * blockDim.x + threadIdx.x;
    if (k >= SDT_NNODES) { if (k < SDT_NPAD1) pencol[k] = 0.0f; return; }
    float num = 0.0f, den = 0.0f;
    for (int b = 0; b < SDT_BATCH; ++b) {                // sequential -> deterministic
        float m = allpp[(size_t)b * SDT_NNODES + k];
        float p = P[(size_t)b * ldp + k];
        num += p * m;
        den += m;
    }
    float alpha = num / den;
    float pen = logf(alpha) + logf(1.0f - alpha);
    if (!__builtin_isfinite(pen)) pen = 0.0f;
    int layer = 31 - __clz(k + 1);                       // floor(log2(k+1))
    pencol[k] = -0.5f * exp2f(-(float)layer) * pen;
}

__global__ __launch_bounds__(256)
void sdt_pen_reduce(const float* __restrict__ pencol, float* __restrict__ out) {
    __shared__ float s[256];
    float acc = 0.0f;
    for (int k = threadIdx.x; k < SDT_NNODES; k += 256) acc += pencol[k];
    s[threadIdx.x] = acc;
    __syncthreads();
    for (int w = 128; w > 0; w >>= 1) {
        if (threadIdx.x < w) s[threadIdx.x] += s[threadIdx.x + w];
        __syncthreads();
    }
    if (threadIdx.x == 0) *out = s[0];
}

// ---------------------- launcher -------------------------------------------

extern "C" void kernel_launch(void* const* d_in, const int* in_sizes, int n_in,
                              void* d_out, int out_size, void* d_ws, size_t ws_size,
                              hipStream_t stream) {
    const float* X  = (const float*)d_in[0];   // 8192 x 2048
    const float* Wi = (const float*)d_in[1];   // 2047 x 2049
    const float* Wl = (const float*)d_in[2];   // 1000 x 1024

    // workspace layout (bytes)
    char* ws = (char*)d_ws;
    bf16*  Xbf    = (bf16*)(ws);                           //  33,554,432
    bf16*  Wt     = (bf16*)(ws + 33554432);                //   8,388,608
    float* bias   = (float*)(ws + 41943040);               //       8,192
    float* P      = (float*)(ws + 41951232);               //  67,108,864 (ld 2048)
    bf16*  MUbf   = (bf16*)(ws + 109060096);               //  16,777,216
    bf16*  WlBf   = (bf16*)(ws + 125837312);               //   2,097,152
    float* pencol = (float*)(ws + 127934464);              //       8,192

    // output layout: logits | mu | penalty | all_path_probas
    float* logits = (float*)d_out;
    float* muOut  = logits + (size_t)SDT_BATCH * SDT_OUTDIM;
    float* penOut = muOut + (size_t)SDT_BATCH * SDT_NLEAF;
    float* allpp  = penOut + 1;

    // 1) precision conversion (one pass, halves GEMM read traffic)
    sdt_conv_x<<<(SDT_BATCH * SDT_INDIM + 255) / 256, 256, 0, stream>>>(
        X, Xbf, SDT_BATCH * SDT_INDIM);
    sdt_conv_winner<<<(SDT_NPAD1 * SDT_KPAD1) / 256, 256, 0, stream>>>(Wi, Wt, bias);
    sdt_conv_wleaf<<<(SDT_NPAD2 * SDT_NLEAF) / 256, 256, 0, stream>>>(Wl, WlBf);

    // 2) GEMM1: P = sigmoid(X @ W_inner[:,1:]^T + bias)   (8192 x 2047)
    sdt_gemm_bf16<true><<<dim3(SDT_NPAD1 / 128, SDT_BATCH / 64), 256, 0, stream>>>(
        Xbf, Wt, bias, P, SDT_KPAD1, 2048, SDT_NNODES);

    // 3) path-probability cascade (-> all_pp, mu, mu_bf16)
    sdt_mu<<<SDT_BATCH, 128, 0, stream>>>(P, 2048, allpp, muOut, MUbf);

    // 4) penalty (deterministic)
    sdt_pen_cols<<<SDT_NPAD1 / 256, 256, 0, stream>>>(P, 2048, allpp, pencol);
    sdt_pen_reduce<<<1, 256, 0, stream>>>(pencol, penOut);

    // 5) GEMM2: logits = mu @ W_leaf^T   (8192 x 1000)
    sdt_gemm_bf16<false><<<dim3(SDT_NPAD2 / 128, SDT_BATCH / 64), 256, 0, stream>>>(
        MUbf, WlBf, nullptr, logits, SDT_NLEAF, SDT_OUTDIM, SDT_OUTDIM);
}
